// HumaniflowModel_30906584662049
// MI455X (gfx1250) — compile-verified, matrix operand order checked by reading, not run on hardware
//
#include <hip/hip_runtime.h>
#include <hip/hip_bf16.h>

typedef __attribute__((ext_vector_type(16))) _Float16 v16h;
typedef __attribute__((ext_vector_type(8)))  _Float16 v8h;
typedef __attribute__((ext_vector_type(4)))  _Float16 v4h;
typedef __attribute__((ext_vector_type(8)))  float    v8f;

#define B_SZ  16384
#define S_SZ  256
#define D_IN  2048
#define D_FC1 1024
#define NB    10

// ---------------- GEMM1: elu(X @ W1 + b1) -> f16 activations ----------------
// Tile: BM=128, BN=128, BK=32. 256 threads = 8 waves; waves in 4(m) x 2(n),
// each wave computes 32x64 = 2x4 grid of 16x16 WMMA tiles.
#define BM 128
#define BN 128
#define BK 32
#define LDA 56   // halves per LDS row: 112 B = 7*16B (aligned, conflict-free)
#define LDB 56

__global__ __launch_bounds__(256)
void gemm1_elu_kernel(const float* __restrict__ X, const float* __restrict__ W,
                      const float* __restrict__ bias, _Float16* __restrict__ xact)
{
    __shared__ _Float16 sA[BM][LDA];      // [m][k]
    __shared__ _Float16 sB[BN][LDB];      // transposed: [n][k]

    const int tid  = threadIdx.x;
    const int lane = tid & 31;
    const int wave = tid >> 5;
    const int wm   = wave & 3;            // 0..3 : 32-row strip
    const int wn   = wave >> 2;           // 0..1 : 64-col strip
    const int l15  = lane & 15;
    const int kg   = lane >> 4;           // K-phase select per ISA A/B layout
    const int gm0  = blockIdx.y * BM;
    const int gn0  = blockIdx.x * BN;

    v8f acc[2][4];
#pragma unroll
    for (int mt = 0; mt < 2; ++mt)
#pragma unroll
        for (int nt = 0; nt < 4; ++nt)
#pragma unroll
            for (int e = 0; e < 8; ++e) acc[mt][nt][e] = 0.0f;

    for (int kb = 0; kb < D_IN; kb += BK) {
        // Stage A: 128x32 f32 -> f16 LDS (1024 float4, 4 per thread)
#pragma unroll
        for (int i = 0; i < 4; ++i) {
            const int idx = tid + i * 256;
            const int r   = idx >> 3;           // 0..127
            const int c4  = idx & 7;            // 0..7 (float4 within row)
            const float4 v = *(const float4*)&X[(size_t)(gm0 + r) * D_IN + kb + c4 * 4];
            v4h h;
            h[0] = (_Float16)v.x; h[1] = (_Float16)v.y;
            h[2] = (_Float16)v.z; h[3] = (_Float16)v.w;
            *(v4h*)&sA[r][c4 * 4] = h;
        }
        // Stage B transposed: 32x128 f32 -> sB[n][k] f16
#pragma unroll
        for (int i = 0; i < 4; ++i) {
            const int idx = tid + i * 256;
            const int kr  = idx >> 5;           // 0..31
            const int c4  = idx & 31;           // n-group
            const float4 v = *(const float4*)&W[(size_t)(kb + kr) * D_FC1 + gn0 + c4 * 4];
            sB[c4 * 4 + 0][kr] = (_Float16)v.x;
            sB[c4 * 4 + 1][kr] = (_Float16)v.y;
            sB[c4 * 4 + 2][kr] = (_Float16)v.z;
            sB[c4 * 4 + 3][kr] = (_Float16)v.w;
        }
        __syncthreads();

        // Build fragments per ISA 16-bit A layout:
        // lane(0-15)=row, lane[4]=K phase; halves 0..7 -> K+0..7, 8..15 -> K+16..23
        v16h af[2], bf[4];
#pragma unroll
        for (int mt = 0; mt < 2; ++mt) {
            const int row = wm * 32 + mt * 16 + l15;
            const v8h lo = *(const v8h*)&sA[row][kg * 8];
            const v8h hi = *(const v8h*)&sA[row][kg * 8 + 16];
#pragma unroll
            for (int e = 0; e < 8; ++e) { af[mt][e] = lo[e]; af[mt][8 + e] = hi[e]; }
        }
#pragma unroll
        for (int nt = 0; nt < 4; ++nt) {
            const int col = wn * 64 + nt * 16 + l15;
            const v8h lo = *(const v8h*)&sB[col][kg * 8];
            const v8h hi = *(const v8h*)&sB[col][kg * 8 + 16];
#pragma unroll
            for (int e = 0; e < 8; ++e) { bf[nt][e] = lo[e]; bf[nt][8 + e] = hi[e]; }
        }
#pragma unroll
        for (int mt = 0; mt < 2; ++mt)
#pragma unroll
            for (int nt = 0; nt < 4; ++nt)
                acc[mt][nt] = __builtin_amdgcn_wmma_f32_16x16x32_f16(
                    false, af[mt], false, bf[nt], (short)0, acc[mt][nt], false, false);
        __syncthreads();
    }

    // Epilogue: + b1, ELU, store f16 activation. C layout: VGPR r -> M = r + 8*lane[4].
#pragma unroll
    for (int mt = 0; mt < 2; ++mt)
#pragma unroll
        for (int nt = 0; nt < 4; ++nt) {
            const int col = gn0 + wn * 64 + nt * 16 + l15;
            const float bv = bias[col];
#pragma unroll
            for (int r = 0; r < 8; ++r) {
                const int row = gm0 + wm * 32 + mt * 16 + (kg << 3) + r;
                float v = acc[mt][nt][r] + bv;
                v = (v > 0.0f) ? v : expm1f(v);
                xact[(size_t)row * D_FC1 + col] = (_Float16)v;
            }
        }
}

// ---------------- GEMM2: xact @ w_shape + b_shape -> mode / log_std ----------------
// N padded 20 -> 32. 8 waves x 16 rows = 128 rows per block. B staged in LDS
// (transposed, f16) in two 512-K chunks (stride 520 halves = 1040B = 65*16B).
#define LDB2 520

__global__ __launch_bounds__(256)
void gemm2_kernel(const _Float16* __restrict__ xact, const float* __restrict__ Wsh,
                  const float* __restrict__ bsh, float* __restrict__ out)
{
    __shared__ _Float16 sB[32][LDB2];     // [n][k-chunk]

    const int tid   = threadIdx.x;
    const int lane  = tid & 31;
    const int wave  = tid >> 5;
    const int l15   = lane & 15;
    const int kg    = lane >> 4;
    const int rbase = blockIdx.x * 128 + wave * 16;

    v8f acc[2];
#pragma unroll
    for (int nt = 0; nt < 2; ++nt)
#pragma unroll
        for (int e = 0; e < 8; ++e) acc[nt][e] = 0.0f;

    for (int kc = 0; kc < D_FC1; kc += 512) {
        // Stage B chunk: 32 cols x 512 k (pad cols >= 20 with zeros)
#pragma unroll 4
        for (int i = 0; i < 64; ++i) {
            const int idx = tid + i * 256;
            const int n   = idx & 31;
            const int k   = idx >> 5;          // 0..511
            const float v = (n < 2 * NB) ? Wsh[(size_t)(kc + k) * (2 * NB) + n] : 0.0f;
            sB[n][k] = (_Float16)v;
        }
        __syncthreads();

        for (int kk = 0; kk < 512; kk += 32) {
            // A fragment straight from global f16 workspace (layout matches)
            const _Float16* ap = &xact[(size_t)(rbase + l15) * D_FC1 + kc + kk + kg * 8];
            const v8h alo = *(const v8h*)ap;
            const v8h ahi = *(const v8h*)(ap + 16);
            v16h a;
#pragma unroll
            for (int e = 0; e < 8; ++e) { a[e] = alo[e]; a[8 + e] = ahi[e]; }
#pragma unroll
            for (int nt = 0; nt < 2; ++nt) {
                const v8h blo = *(const v8h*)&sB[nt * 16 + l15][kk + kg * 8];
                const v8h bhi = *(const v8h*)&sB[nt * 16 + l15][kk + kg * 8 + 16];
                v16h b;
#pragma unroll
                for (int e = 0; e < 8; ++e) { b[e] = blo[e]; b[8 + e] = bhi[e]; }
                acc[nt] = __builtin_amdgcn_wmma_f32_16x16x32_f16(
                    false, a, false, b, (short)0, acc[nt], false, false);
            }
        }
        __syncthreads();
    }

    // Epilogue: split cols into mode (0..9) and log_std (10..19) slices of d_out
#pragma unroll
    for (int nt = 0; nt < 2; ++nt) {
        const int c = nt * 16 + l15;
        if (c < 2 * NB) {
            const float bv = bsh[c];
#pragma unroll
            for (int r = 0; r < 8; ++r) {
                const int row = rbase + (kg << 3) + r;
                const float v = acc[nt][r] + bv;
                if (c < NB) out[(size_t)row * NB + c] = v;
                else        out[(size_t)B_SZ * NB + (size_t)row * NB + (c - NB)] = v;
            }
        }
    }
}

// ---------------- Sampling: mode + exp(log_std) * noise ----------------
__global__ __launch_bounds__(256)
void sample_kernel(const float* __restrict__ noise, const float* __restrict__ mode,
                   const float* __restrict__ lstd, float* __restrict__ outS)
{
    __shared__ float m[NB];
    __shared__ float es[NB];
    const int b = blockIdx.x;
    if (threadIdx.x < NB) {
        m[threadIdx.x]  = mode[(size_t)b * NB + threadIdx.x];
        es[threadIdx.x] = expf(lstd[(size_t)b * NB + threadIdx.x]);
    }
    __syncthreads();

    const int total = S_SZ * NB;               // 2560
    const float* np = noise + (size_t)b * total;
    float*       op = outS  + (size_t)b * total;
    for (int e = threadIdx.x; e < total; e += 256) {
        const int j = e % NB;
        op[e] = m[j] + es[j] * np[e];
    }
}

// ---------------- launch ----------------
extern "C" void kernel_launch(void* const* d_in, const int* in_sizes, int n_in,
                              void* d_out, int out_size, void* d_ws, size_t ws_size,
                              hipStream_t stream)
{
    const float* X     = (const float*)d_in[0];   // (16384, 2048)
    const float* W1    = (const float*)d_in[1];   // (2048, 1024)
    const float* b1    = (const float*)d_in[2];   // (1024,)
    const float* Wsh   = (const float*)d_in[3];   // (1024, 20)
    const float* bsh   = (const float*)d_in[4];   // (20,)
    const float* noise = (const float*)d_in[5];   // (16384, 256, 10)
    // d_in[6] = num_samples (compile-time constant S_SZ here)

    float* out = (float*)d_out;                   // [mode | log_std | samples]
    _Float16* xact = (_Float16*)d_ws;             // 16384*1024 f16 = 32 MiB

    dim3 g1(D_FC1 / BN, B_SZ / BM);               // (8, 128)
    gemm1_elu_kernel<<<g1, 256, 0, stream>>>(X, W1, b1, xact);

    gemm2_kernel<<<dim3(B_SZ / 128), 256, 0, stream>>>(xact, Wsh, bsh, out);

    sample_kernel<<<dim3(B_SZ), 256, 0, stream>>>(
        noise, out, out + (size_t)B_SZ * NB, out + 2 * (size_t)B_SZ * NB);
}
